// SAmodule_20246475833546
// MI455X (gfx1250) — matrix operand order repaired and checked by reading, not compiled, not run on hardware
//
#include <hip/hip_runtime.h>
#include <hip/hip_bf16.h>

// Self-attention (SAGAN) for MI455X / gfx1250, wave32 + bf16 WMMA.
// Sizes fixed by the reference: B=4, C=256, HW=64*64=4096, c=32.
#define HW   4096
#define CIN  256
#define CMID 32
#define NB   4

typedef __attribute__((ext_vector_type(16))) __bf16  v16bf;
typedef __attribute__((ext_vector_type(8)))  float   v8f;
typedef __attribute__((ext_vector_type(4)))  unsigned u32x4;

union ABu { v16bf v; u32x4 q[2]; unsigned u[8]; };

__device__ __forceinline__ unsigned short f2bf(float f) {
  unsigned u = __float_as_uint(f);
  return (unsigned short)((u + 0x7FFFu + ((u >> 16) & 1u)) >> 16);  // RNE
}
__device__ __forceinline__ unsigned pack2(float a, float b) {
  return (unsigned)f2bf(a) | ((unsigned)f2bf(b) << 16);
}
__device__ __forceinline__ v8f wmma_bf16(const ABu& a, const ABu& b, v8f c) {
  // D = A(16x32 bf16) * B(32x16 bf16) + C(16x16 f32)
  return __builtin_amdgcn_wmma_f32_16x16x32_bf16(false, a.v, false, b.v,
                                                 (short)0, c, false, false);
}

// ---------------------------------------------------------------------------
// Kernel 1: 1x1-conv projections. blockIdx.z selects f/g/h.
//   ft[b][q][c] = f (bf16, c contiguous  -> WMMA A layout for scores)
//   gt[b][k][c] = g (bf16, c contiguous  -> WMMA B layout for scores)
//   ht[b][c][q] = h (bf16, q contiguous  -> WMMA A layout for o-GEMM)
// ---------------------------------------------------------------------------
__global__ __launch_bounds__(256) void proj_kernel(
    const float* __restrict__ x,
    const float* __restrict__ w1, const float* __restrict__ b1,
    const float* __restrict__ w2, const float* __restrict__ b2,
    const float* __restrict__ w3, const float* __restrict__ b3,
    __bf16* __restrict__ ft, __bf16* __restrict__ gt, __bf16* __restrict__ ht)
{
  __shared__ float ws[CMID * CIN];            // 32 KB of the 320 KB WGP LDS
  const int which = blockIdx.z;
  const float* w    = (which == 0) ? w1 : (which == 1) ? w2 : w3;
  const float* bias = (which == 0) ? b1 : (which == 1) ? b2 : b3;
  for (int i = threadIdx.x; i < CMID * CIN; i += 256) ws[i] = w[i];
  __syncthreads();

  const int q = blockIdx.x * 256 + threadIdx.x;   // coalesced over q
  const int b = blockIdx.y;

  float acc[CMID];
  #pragma unroll
  for (int cc = 0; cc < CMID; ++cc) acc[cc] = bias[cc];

  const float* xp = x + ((size_t)b * CIN) * HW + q;
  for (int C = 0; C < CIN; ++C) {
    float xv = xp[(size_t)C * HW];
    #pragma unroll
    for (int cc = 0; cc < CMID; ++cc)
      acc[cc] = fmaf(ws[cc * CIN + C], xv, acc[cc]);
  }

  if (which < 2) {
    __bf16* row = (which == 0 ? ft : gt) + ((size_t)b * HW + q) * CMID;
    u32x4* dst = reinterpret_cast<u32x4*>(row);
    #pragma unroll
    for (int i = 0; i < 4; ++i) {
      u32x4 t;
      t[0] = pack2(acc[8*i+0], acc[8*i+1]);
      t[1] = pack2(acc[8*i+2], acc[8*i+3]);
      t[2] = pack2(acc[8*i+4], acc[8*i+5]);
      t[3] = pack2(acc[8*i+6], acc[8*i+7]);
      dst[i] = t;
    }
  } else {
    unsigned short* hs = reinterpret_cast<unsigned short*>(ht);
    #pragma unroll
    for (int cc = 0; cc < CMID; ++cc)
      hs[((size_t)b * CMID + cc) * HW + q] = f2bf(acc[cc]);  // coalesced over q
  }
}

// ---------------------------------------------------------------------------
// Kernel 2: fused attention (flash over the q axis, since softmax is axis=1).
// One wave owns 16 k-columns: Bg (g[32, k0:k0+16]) is held in regs for the
// whole kernel. Loop q in blocks of 32:
//   scores: 2x v_wmma_f32_16x16x32_bf16  (K = c = 32 exactly)
//   online max/sum per lane (each lane owns one k; lanes L and L+16 pair up
//   via shfl_xor(16) to cover all 32 q of a block)
//   p -> bf16, recombined into the B layout of the o-GEMM with 8 shfl_xor
//   o-acc: 2x v_wmma_f32_16x16x32_bf16  (c rows 0..15 and 16..31)
// ---------------------------------------------------------------------------
__global__ __launch_bounds__(128) void attn_kernel(
    const __bf16* __restrict__ ft, const __bf16* __restrict__ gt,
    const __bf16* __restrict__ ht, float* __restrict__ o)
{
  const int lane = threadIdx.x & 31;
  const int wave = threadIdx.x >> 5;
  const int b    = blockIdx.y;
  const int k0   = (blockIdx.x * 4 + wave) * 16;
  const bool lowhalf = lane < 16;
  const int l15 = lane & 15;
  const int hi8 = (lane >> 4) << 3;          // 0 for lanes 0-15, 8 for 16-31

  // B operand (scores): g[c, k0+n]; lane<16: n=lane, K=c 0..7 and 16..23;
  // lane>=16: n=lane-16, K=c 8..15 and 24..31.
  ABu Bg;
  {
    const __bf16* gp = gt + ((size_t)b * HW + k0 + l15) * CMID + hi8;
    Bg.q[0] = *reinterpret_cast<const u32x4*>(gp);
    Bg.q[1] = *reinterpret_cast<const u32x4*>(gp + 16);
  }

  const __bf16* flane  = ft + (size_t)b * HW * CMID + (size_t)l15 * CMID + hi8;
  const __bf16* hlane0 = ht + (size_t)b * CMID * HW + (size_t)l15 * HW + (hi8 << 1);
  const __bf16* hlane1 = hlane0 + (size_t)16 * HW;

  v8f o0 = {0,0,0,0,0,0,0,0};
  v8f o1 = {0,0,0,0,0,0,0,0};
  float m = -INFINITY, l = 0.0f;

  for (int q0 = 0; q0 < HW; q0 += 32) {
    // A (scores): f^T rows q0..q0+31 ; two 16-row tiles.
    ABu Af0, Af1, Ah0, Ah1;
    const __bf16* fp = flane + (size_t)q0 * CMID;
    Af0.q[0] = *reinterpret_cast<const u32x4*>(fp);
    Af0.q[1] = *reinterpret_cast<const u32x4*>(fp + 16);
    Af1.q[0] = *reinterpret_cast<const u32x4*>(fp + 16 * CMID);
    Af1.q[1] = *reinterpret_cast<const u32x4*>(fp + 16 * CMID + 16);
    // A (o-GEMM): h rows c 0..15 and 16..31, K = q0..q0+31.
    const __bf16* hp0 = hlane0 + q0;
    Ah0.q[0] = *reinterpret_cast<const u32x4*>(hp0);
    Ah0.q[1] = *reinterpret_cast<const u32x4*>(hp0 + 8);
    const __bf16* hp1 = hlane1 + q0;
    Ah1.q[0] = *reinterpret_cast<const u32x4*>(hp1);
    Ah1.q[1] = *reinterpret_cast<const u32x4*>(hp1 + 8);

    v8f z = {0,0,0,0,0,0,0,0};
    v8f s0 = wmma_bf16(Af0, Bg, z);   // q = q0 + (lowhalf? r : 8+r)
    v8f s1 = wmma_bf16(Af1, Bg, z);   // q = q0+16 + (lowhalf? r : 8+r)

    // Online softmax over q for this lane's k = k0 + l15.
    float bm = s0[0];
    #pragma unroll
    for (int i = 1; i < 8; ++i) bm = fmaxf(bm, s0[i]);
    #pragma unroll
    for (int i = 0; i < 8; ++i) bm = fmaxf(bm, s1[i]);
    bm = fmaxf(bm, __shfl_xor(bm, 16));           // pair lanes L <-> L+16
    const float nm    = fmaxf(m, bm);
    const float scale = __expf(m - nm);           // exp(-inf)=0 on first block
    float p0[8], p1[8], bs = 0.0f;
    #pragma unroll
    for (int i = 0; i < 8; ++i) { p0[i] = __expf(s0[i] - nm); bs += p0[i]; }
    #pragma unroll
    for (int i = 0; i < 8; ++i) { p1[i] = __expf(s1[i] - nm); bs += p1[i]; }
    bs += __shfl_xor(bs, 16);
    l = l * scale + bs;
    m = nm;
    #pragma unroll
    for (int i = 0; i < 8; ++i) { o0[i] *= scale; o1[i] *= scale; }

    // Build B operand p[q, k] (32x16 bf16). Lane<16 needs q0+0..15:
    // own tile0 half + partner's tile0 half. Lane>=16 needs q0+16..31:
    // partner's tile1 half + own tile1 half. 8 shfl_xor total.
    unsigned t0[4], t1[4];
    #pragma unroll
    for (int i = 0; i < 4; ++i) {
      t0[i] = pack2(p0[2*i], p0[2*i+1]);
      t1[i] = pack2(p1[2*i], p1[2*i+1]);
    }
    ABu Bp;
    #pragma unroll
    for (int i = 0; i < 4; ++i) {
      const unsigned r0 = __shfl_xor(t0[i], 16);
      const unsigned r1 = __shfl_xor(t1[i], 16);
      Bp.u[i]     = lowhalf ? t0[i] : r1;
      Bp.u[4 + i] = lowhalf ? r0    : t1[i];
    }

    o0 = wmma_bf16(Ah0, Bp, o0);      // c rows 0..15
    o1 = wmma_bf16(Ah1, Bp, o1);      // c rows 16..31
  }

  // Normalize by l and store o[b][c][k] (f32, k contiguous).
  const float inv = 1.0f / l;
  const int kk = k0 + l15;
  float* ob = o + (size_t)b * CMID * HW + kk;
  #pragma unroll
  for (int r = 0; r < 8; ++r)
    ob[(size_t)(hi8 + r) * HW] = o0[r] * inv;        // c = hi8 + r
  #pragma unroll
  for (int r = 0; r < 8; ++r)
    ob[(size_t)(16 + hi8 + r) * HW] = o1[r] * inv;   // c = 16 + hi8 + r
}

// ---------------------------------------------------------------------------
// Kernel 3: out = w4 . o + b4 + x   (tiny: 0.27 GFLOP, bandwidth-bound)
// ---------------------------------------------------------------------------
__global__ __launch_bounds__(256) void out_kernel(
    const float* __restrict__ x,  const float* __restrict__ w4,
    const float* __restrict__ b4, const float* __restrict__ o,
    float* __restrict__ out)
{
  const int q  = blockIdx.x * 256 + threadIdx.x;
  const int Co = blockIdx.y;
  const int b  = blockIdx.z;
  float acc = b4[Co] + x[((size_t)b * CIN + Co) * HW + q];
  const float* wrow = w4 + (size_t)Co * CMID;          // uniform -> scalar
  const float* orow = o  + (size_t)b * CMID * HW + q;  // coalesced over q
  #pragma unroll
  for (int c = 0; c < CMID; ++c)
    acc = fmaf(wrow[c], orow[(size_t)c * HW], acc);
  out[((size_t)b * CIN + Co) * HW + q] = acc;
}

// ---------------------------------------------------------------------------
extern "C" void kernel_launch(void* const* d_in, const int* in_sizes, int n_in,
                              void* d_out, int out_size, void* d_ws, size_t ws_size,
                              hipStream_t stream) {
  const float* x  = (const float*)d_in[0];
  const float* w1 = (const float*)d_in[1];
  const float* b1 = (const float*)d_in[2];
  const float* w2 = (const float*)d_in[3];
  const float* b2 = (const float*)d_in[4];
  const float* w3 = (const float*)d_in[5];
  const float* b3 = (const float*)d_in[6];
  const float* w4 = (const float*)d_in[7];
  const float* b4 = (const float*)d_in[8];
  float* out = (float*)d_out;

  // Workspace layout (5 MB total):
  //   [0,1MB)  ft bf16 [B][HW][32]
  //   [1,2MB)  gt bf16 [B][HW][32]
  //   [2,3MB)  ht bf16 [B][32][HW]
  //   [3,5MB)  o  f32  [B][32][HW]
  char* ws = (char*)d_ws;
  __bf16* ft = (__bf16*)(ws);
  __bf16* gt = (__bf16*)(ws + (size_t)(1u << 20));
  __bf16* ht = (__bf16*)(ws + (size_t)(2u << 20));
  float*  o  = (float*) (ws + (size_t)(3u << 20));

  proj_kernel<<<dim3(HW / 256, NB, 3), 256, 0, stream>>>(
      x, w1, b1, w2, b2, w3, b3, ft, gt, ht);
  attn_kernel<<<dim3(HW / 64, NB), 128, 0, stream>>>(ft, gt, ht, o);
  out_kernel<<<dim3(HW / 256, CIN, NB), 256, 0, stream>>>(x, w4, b4, o, out);
}
